// AttnBlock_12807592477253
// MI455X (gfx1250) — matrix operand order repaired and checked by reading, not compiled, not run on hardware
//
#include <hip/hip_runtime.h>
#include <cstdint>
#include <cfloat>
#include <math.h>

typedef __attribute__((ext_vector_type(16))) __bf16 bf16x16;
typedef __attribute__((ext_vector_type(8)))  float  f32x8;
typedef __attribute__((ext_vector_type(4)))  float  f32x4;
typedef __attribute__((ext_vector_type(4)))  unsigned int u32x4;
typedef __attribute__((ext_vector_type(4)))  int    i32x4v;

#define DEV __device__ __forceinline__

// ---------------- CDNA5 async global->LDS (guarded), sync fallback ----------
#if defined(__HIP_DEVICE_COMPILE__)
  #if defined(__has_builtin)
    #if __has_builtin(__builtin_amdgcn_global_load_async_to_lds_b128) && \
        __has_builtin(__builtin_amdgcn_s_wait_asynccnt)
      #define USE_ASYNC_LDS 1
    #endif
  #endif
#endif

#if defined(USE_ASYNC_LDS)
typedef __attribute__((address_space(1))) i32x4v ga_i32x4;
typedef __attribute__((address_space(3))) i32x4v ls_i32x4;
#endif

DEV void cp16(const void* g, void* l) {
#if defined(USE_ASYNC_LDS)
  __builtin_amdgcn_global_load_async_to_lds_b128(
      (ga_i32x4*)(uintptr_t)g, (ls_i32x4*)(uintptr_t)l, 0, 0);
#else
  *(u32x4*)l = *(const u32x4*)g;
#endif
}
DEV void cp_wait_newer4() {
#if defined(USE_ASYNC_LDS)
  __builtin_amdgcn_s_wait_asynccnt(4);
#endif
}
DEV void cp_wait_all() {
#if defined(USE_ASYNC_LDS)
  __builtin_amdgcn_s_wait_asynccnt(0);
#endif
}

// ---------------- fragment helpers (CDNA5 16x16x32 bf16 layouts) ------------
union FragB16 { bf16x16 v; u32x4 q[2]; __bf16 h[16]; };

// A (16x32): lane m = lane&15 is the row; elements 0..7 -> K = (lane>>4)*8+e,
// elements 8..15 -> K = 16+(lane>>4)*8+(e-8).
DEV bf16x16 ld_fragA(const __bf16* rowp, int lane) {
  const __bf16* p = rowp + ((lane >> 4) << 3);
  FragB16 f;
  f.q[0] = *(const u32x4*)p;
  f.q[1] = *(const u32x4*)(p + 16);
  return f.v;
}
// B (32x16): lane n = lane&15 is the column; elements 0..7 -> K = (lane>>4)*8+e,
// elements 8..15 -> K = 16+(lane>>4)*8+(e-8).
DEV bf16x16 ld_fragB(const __bf16* colp, int lane) {
  const __bf16* p = colp + ((lane >> 4) << 3);
  FragB16 f;
  f.q[0] = *(const u32x4*)p;
  f.q[1] = *(const u32x4*)(p + 16);
  return f.v;
}

DEV f32x8 zero8() {
  f32x8 z;
#pragma unroll
  for (int i = 0; i < 8; ++i) z[i] = 0.0f;
  return z;
}

DEV f32x8 wmma_bf16(bf16x16 a, bf16x16 b, f32x8 c) {
  return __builtin_amdgcn_wmma_f32_16x16x32_bf16(false, a, false, b,
                                                 (short)0, c, false, false);
}

DEV float gelu_tanh(float v) {
  float u = 0.7978845608028654f * (v + 0.044715f * v * v * v);
  return 0.5f * v * (1.0f + tanhf(u));
}

// ---------------- weight convert + transpose: wt[n*K+k] = (bf16)w[k*Nc+n] ----
__global__ void __launch_bounds__(256) cvt_wT_kernel(const float* __restrict__ w,
                                                     __bf16* __restrict__ wt,
                                                     int K, int Nc) {
  int idx = blockIdx.x * 256 + threadIdx.x;  // over K*Nc
  int n = idx / K;
  int k = idx - n * K;
  wt[idx] = (__bf16)w[(size_t)k * Nc + n];
}

// ---------------- LayerNorm (C=1024), f32 in -> bf16 out --------------------
__global__ void __launch_bounds__(256) ln_kernel(const float* __restrict__ x,
                                                 __bf16* __restrict__ out) {
  const int C = 1024;
  const int row = blockIdx.x, tid = threadIdx.x;
  const float* xr = x + (size_t)row * C;
  f32x4 v = *(const f32x4*)(xr + tid * 4);
  float s = v[0] + v[1] + v[2] + v[3];
  float s2 = v[0]*v[0] + v[1]*v[1] + v[2]*v[2] + v[3]*v[3];
  __shared__ float r1[256], r2[256];
  r1[tid] = s; r2[tid] = s2;
  __syncthreads();
  for (int off = 128; off > 0; off >>= 1) {
    if (tid < off) { r1[tid] += r1[tid + off]; r2[tid] += r2[tid + off]; }
    __syncthreads();
  }
  float mu   = r1[0] * (1.0f / 1024.0f);
  float var  = r2[0] * (1.0f / 1024.0f) - mu * mu;
  float rstd = 1.0f / sqrtf(var + 1e-6f);
  __bf16* orow = out + (size_t)row * C + tid * 4;
#pragma unroll
  for (int j = 0; j < 4; ++j) orow[j] = (__bf16)((v[j] - mu) * rstd);
}

// ---------------- generic bf16 WMMA GEMM ------------------------------------
// C[M,Nc] = act(A[M,K] @ Bt[Nc,K]^T + bias) (+ res), out bf16 and/or f32.
// Block tile 128x128, K-tile 32, 8 waves each computing 64x32.
template<int ACT, bool HAS_RES, bool OUT_BF16>
__global__ void __launch_bounds__(256) gemm_bf16(
    const __bf16* __restrict__ A, const __bf16* __restrict__ Bt,
    const float* __restrict__ bias, const float* __restrict__ res,
    __bf16* __restrict__ outb, float* __restrict__ outf,
    int M, int Nc, int K)
{
  constexpr int BK = 32, LDT = 40;             // LDS row stride in halves
  constexpr int TILE_BYTES = 128 * LDT * 2;    // 10240 B per matrix tile
  __shared__ __align__(16) char lds[2 * 2 * TILE_BYTES];  // double buffered A+B

  const int tid = threadIdx.x;
  const int lane = tid & 31, wid = tid >> 5;
  const int wm = wid & 1, wn = wid >> 1;
  const int rowBase = blockIdx.y * 128, colBase = blockIdx.x * 128;

  auto issue = [&](int kt, int buf) {
    char* ab = lds + buf * (2 * TILE_BYTES);
    char* bb = ab + TILE_BYTES;
    const int kb = kt * BK;
#pragma unroll
    for (int j = 0; j < 2; ++j) {             // 512 16B chunks of A tile
      int l = tid + j * 256;
      int r = l >> 2, cg = (l & 3) * 8;
      cp16(A + (size_t)(rowBase + r) * K + kb + cg, ab + (r * LDT + cg) * 2);
    }
#pragma unroll
    for (int j = 0; j < 2; ++j) {             // 512 16B chunks of B tile
      int l = tid + j * 256;
      int r = l >> 2, cg = (l & 3) * 8;
      cp16(Bt + (size_t)(colBase + r) * K + kb + cg, bb + (r * LDT + cg) * 2);
    }
  };

  f32x8 acc[4][2];
#pragma unroll
  for (int mi = 0; mi < 4; ++mi)
#pragma unroll
    for (int ni = 0; ni < 2; ++ni) acc[mi][ni] = zero8();

  const int KT = K / BK;
  issue(0, 0);
  if (KT > 1) issue(1, 1);

  for (int kt = 0; kt < KT; ++kt) {
    const int cur = kt & 1;
    if (kt + 1 < KT) cp_wait_newer4(); else cp_wait_all();
    __syncthreads();

    const __bf16* as = (const __bf16*)(lds + cur * 2 * TILE_BYTES);
    const __bf16* bs = (const __bf16*)(lds + cur * 2 * TILE_BYTES + TILE_BYTES);

    bf16x16 afr[4], bfr[2];
#pragma unroll
    for (int mi = 0; mi < 4; ++mi)
      afr[mi] = ld_fragA(as + (wm * 64 + mi * 16 + (lane & 15)) * LDT, lane);
#pragma unroll
    for (int ni = 0; ni < 2; ++ni)
      bfr[ni] = ld_fragB(bs + (wn * 32 + ni * 16 + (lane & 15)) * LDT, lane);
#pragma unroll
    for (int mi = 0; mi < 4; ++mi)
#pragma unroll
      for (int ni = 0; ni < 2; ++ni)
        acc[mi][ni] = wmma_bf16(afr[mi], bfr[ni], acc[mi][ni]);

    __syncthreads();
    if (kt + 2 < KT) issue(kt + 2, cur);
  }

  // epilogue: C layout: VGPR v, lanes 0-15 -> M=v, lanes 16-31 -> M=v+8; N=lane&15
  const int mrow = (lane >> 4) * 8;
  const int ncol = lane & 15;
#pragma unroll
  for (int mi = 0; mi < 4; ++mi)
#pragma unroll
    for (int ni = 0; ni < 2; ++ni) {
      const int gcol = colBase + wn * 32 + ni * 16 + ncol;
      const float bv = bias[gcol];
#pragma unroll
      for (int v = 0; v < 8; ++v) {
        const int grow = rowBase + wm * 64 + mi * 16 + mrow + v;
        float val = acc[mi][ni][v] + bv;
        if (ACT == 1) val = gelu_tanh(val);
        if (HAS_RES) val += res[(size_t)grow * Nc + gcol];
        if (OUT_BF16) outb[(size_t)grow * Nc + gcol] = (__bf16)val;
        else          outf[(size_t)grow * Nc + gcol] = val;
      }
    }
}

// ---------------- attention: one block = (b,h) x 32 query rows --------------
// Dynamic LDS layout (bytes):
//   S[32][1024] f32     @ 0        131072
//   Vt[64][1024] bf16   @ 131072   131072
//   Qs[32][72] bf16     @ 262144     4608
//   red[32][8] f32      @ 266752     1024
//   rowsum[32] f32      @ 267776      128
//   Mb[32][1024] u8     @ 267904    32768   (mask tile staged in LDS)
#define ATTN_SMEM_BYTES 300672
__global__ void __launch_bounds__(256) attn_kernel(
    const __bf16* __restrict__ qg, const __bf16* __restrict__ kvg,
    const unsigned char* __restrict__ mask, __bf16* __restrict__ og)
{
  constexpr int N = 1024, C = 1024, HD = 64, KVLD = 2048;
  extern __shared__ char smem[];
  float*         S      = (float*)smem;
  __bf16*        Vt     = (__bf16*)(smem + 131072);
  __bf16*        Qs     = (__bf16*)(smem + 262144);
  float*         red    = (float*)(smem + 266752);
  float*         rowsum = (float*)(smem + 267776);
  unsigned char* Mb     = (unsigned char*)(smem + 267904);

  const int tid = threadIdx.x, lane = tid & 31, wid = tid >> 5;
  const int qt = blockIdx.x;            // 32-row query tile
  const int b  = blockIdx.y >> 4, h = blockIdx.y & 15;
  const size_t tokBase = (size_t)b * N;

  // stage Q tile [32][64]
  {
    int r = tid >> 3, cg = (tid & 7) * 8;
    const __bf16* src = qg + (tokBase + qt * 32 + r) * C + h * HD + cg;
    *(u32x4*)(Qs + r * 72 + cg) = *(const u32x4*)src;
  }
  // stage mask tile [32][1024] bytes, coalesced 16B chunks (8 per thread)
  {
    const unsigned char* mrow = mask + (size_t)(qt * 32) * N;
#pragma unroll
    for (int it = 0; it < 8; ++it) {
      int l = tid + it * 256;           // 2048 chunks total
      int r = l >> 6, cg = (l & 63) * 16;
      *(u32x4*)(Mb + r * 1024 + cg) = *(const u32x4*)(mrow + (size_t)r * N + cg);
    }
  }
  // stage V transposed: Vt[d][key]
  {
    const __bf16* vbase = kvg + tokBase * KVLD + C + h * HD;
    for (int it = 0; it < 32; ++it) {
      int l = tid + it * 256;
      int key = l >> 3, dg = (l & 7) * 8;
      union { u32x4 q; __bf16 hh[8]; } u;
      u.q = *(const u32x4*)(vbase + (size_t)key * KVLD + dg);
#pragma unroll
      for (int j = 0; j < 8; ++j) Vt[(dg + j) * 1024 + key] = u.hh[j];
    }
  }
  __syncthreads();

  // ---- phase 1: S = Q K^T * scale + mask bias (wave w owns keys w*128..+128)
  {
    bf16x16 aq[2][2];
#pragma unroll
    for (int mi = 0; mi < 2; ++mi)
#pragma unroll
      for (int ks = 0; ks < 2; ++ks)
        aq[mi][ks] = ld_fragA(Qs + (mi * 16 + (lane & 15)) * 72 + ks * 32, lane);

    const __bf16* kbase = kvg + tokBase * KVLD + h * HD;
    for (int ni = 0; ni < 8; ++ni) {
      const int key16 = wid * 128 + ni * 16;
      bf16x16 bk[2];
#pragma unroll
      for (int ks = 0; ks < 2; ++ks)   // B frag straight from global K rows
        bk[ks] = ld_fragB(kbase + (size_t)(key16 + (lane & 15)) * KVLD + ks * 32, lane);
#pragma unroll
      for (int mi = 0; mi < 2; ++mi) {
        f32x8 acc = zero8();
        acc = wmma_bf16(aq[mi][0], bk[0], acc);
        acc = wmma_bf16(aq[mi][1], bk[1], acc);
#pragma unroll
        for (int v = 0; v < 8; ++v) {
          const int rl  = mi * 16 + (lane >> 4) * 8 + v;
          const int key = key16 + (lane & 15);
          const float bias = Mb[rl * 1024 + key] ? 0.0f : -FLT_MAX;
          S[rl * 1024 + key] = acc[v] * 0.125f + bias;
        }
      }
    }
  }
  __syncthreads();

  // ---- softmax over S rows (8 threads per row), keep unnormalized exp ------
  {
    const int r = tid >> 3, sg = tid & 7;
    float* srow = S + r * 1024 + sg * 128;
    float mx = -FLT_MAX;
    for (int i = 0; i < 128; ++i) mx = fmaxf(mx, srow[i]);
    red[r * 8 + sg] = mx;
    __syncthreads();
    mx = -FLT_MAX;
    for (int j = 0; j < 8; ++j) mx = fmaxf(mx, red[r * 8 + j]);
    __syncthreads();
    float sum = 0.0f;
    for (int i = 0; i < 128; ++i) { float e = __expf(srow[i] - mx); srow[i] = e; sum += e; }
    red[r * 8 + sg] = sum;
    __syncthreads();
    if (sg == 0) {
      float t = 0.0f;
      for (int j = 0; j < 8; ++j) t += red[r * 8 + j];
      rowsum[r] = t;
    }
  }
  __syncthreads();

  // ---- phase 2: O = P V (wave w -> output tile mi=w&1, nj=w>>1) ------------
  {
    const int mi = wid & 1, nj = wid >> 1;
    f32x8 acc = zero8();
    for (int kt = 0; kt < 32; ++kt) {
      const int kb = kt * 32;
      FragB16 fa;
      {
        const float* p0 = S + (mi * 16 + (lane & 15)) * 1024 + kb + (lane >> 4) * 8;
        f32x4 x0 = *(const f32x4*)p0;
        f32x4 x1 = *(const f32x4*)(p0 + 4);
        f32x4 x2 = *(const f32x4*)(p0 + 16);
        f32x4 x3 = *(const f32x4*)(p0 + 20);
#pragma unroll
        for (int j = 0; j < 4; ++j) {
          fa.h[j]      = (__bf16)x0[j];
          fa.h[4 + j]  = (__bf16)x1[j];
          fa.h[8 + j]  = (__bf16)x2[j];
          fa.h[12 + j] = (__bf16)x3[j];
        }
      }
      bf16x16 fb = ld_fragB(Vt + (nj * 16 + (lane & 15)) * 1024 + kb, lane);
      acc = wmma_bf16(fa.v, fb, acc);
    }
#pragma unroll
    for (int v = 0; v < 8; ++v) {
      const int rl = mi * 16 + (lane >> 4) * 8 + v;
      const float val = acc[v] / rowsum[rl];
      og[(tokBase + qt * 32 + rl) * C + h * HD + nj * 16 + (lane & 15)] = (__bf16)val;
    }
  }
}

// ---------------- host orchestration ----------------------------------------
extern "C" void kernel_launch(void* const* d_in, const int* in_sizes, int n_in,
                              void* d_out, int out_size, void* d_ws, size_t ws_size,
                              hipStream_t stream) {
  (void)in_sizes; (void)n_in; (void)out_size; (void)ws_size;
  const int B = 4, N = 1024, C = 1024, Dff = 4096;
  const int M = B * N;

  const float* x            = (const float*)d_in[0];
  const unsigned char* mask = (const unsigned char*)d_in[1];
  const float* wq  = (const float*)d_in[2];
  const float* bq  = (const float*)d_in[3];
  const float* wkv = (const float*)d_in[4];
  const float* bkv = (const float*)d_in[5];
  const float* wo  = (const float*)d_in[6];
  const float* bo  = (const float*)d_in[7];
  const float* w1  = (const float*)d_in[8];
  const float* b1  = (const float*)d_in[9];
  const float* w2  = (const float*)d_in[10];
  const float* b2  = (const float*)d_in[11];
  float* out = (float*)d_out;

  char* p = (char*)d_ws;
  auto take = [&](size_t bytes) { char* r = p; p += (bytes + 255) & ~(size_t)255; return r; };
  __bf16* wqT  = (__bf16*)take((size_t)C * C * 2);
  __bf16* wkvT = (__bf16*)take((size_t)C * 2 * C * 2);
  __bf16* woT  = (__bf16*)take((size_t)C * C * 2);
  __bf16* w1T  = (__bf16*)take((size_t)C * Dff * 2);
  __bf16* w2T  = (__bf16*)take((size_t)Dff * C * 2);
  __bf16* xn   = (__bf16*)take((size_t)M * C * 2);
  __bf16* qb   = (__bf16*)take((size_t)M * C * 2);
  __bf16* kvb  = (__bf16*)take((size_t)M * 2 * C * 2);
  __bf16* ob   = (__bf16*)take((size_t)M * C * 2);
  float*  x1   = (float*) take((size_t)M * C * 4);
  __bf16* xn2  = (__bf16*)take((size_t)M * C * 2);
  __bf16* hb   = (__bf16*)take((size_t)M * Dff * 2);

  // one-time weight converts (transposed to [N][K] bf16)
  cvt_wT_kernel<<<(C * C) / 256,   256, 0, stream>>>(wq,  wqT,  C,   C);
  cvt_wT_kernel<<<(C * 2*C) / 256, 256, 0, stream>>>(wkv, wkvT, C,   2 * C);
  cvt_wT_kernel<<<(C * C) / 256,   256, 0, stream>>>(wo,  woT,  C,   C);
  cvt_wT_kernel<<<(C * Dff) / 256, 256, 0, stream>>>(w1,  w1T,  C,   Dff);
  cvt_wT_kernel<<<(Dff * C) / 256, 256, 0, stream>>>(w2,  w2T,  Dff, C);

  // LN1 -> Q / KV projections
  ln_kernel<<<M, 256, 0, stream>>>(x, xn);
  gemm_bf16<0, false, true><<<dim3(C / 128, M / 128), 256, 0, stream>>>(
      xn, wqT, bq, nullptr, qb, nullptr, M, C, C);
  gemm_bf16<0, false, true><<<dim3(2 * C / 128, M / 128), 256, 0, stream>>>(
      xn, wkvT, bkv, nullptr, kvb, nullptr, M, 2 * C, C);

  // attention
  (void)hipFuncSetAttribute(reinterpret_cast<const void*>(attn_kernel),
                            hipFuncAttributeMaxDynamicSharedMemorySize,
                            ATTN_SMEM_BYTES);
  attn_kernel<<<dim3(N / 32, B * 16), 256, ATTN_SMEM_BYTES, stream>>>(qb, kvb, mask, ob);

  // O-proj + residual (f32), LN2, MLP
  gemm_bf16<0, true, false><<<dim3(C / 128, M / 128), 256, 0, stream>>>(
      ob, woT, bo, x, nullptr, x1, M, C, C);
  ln_kernel<<<M, 256, 0, stream>>>(x1, xn2);
  gemm_bf16<1, false, true><<<dim3(Dff / 128, M / 128), 256, 0, stream>>>(
      xn2, w1T, b1, nullptr, hb, nullptr, M, Dff, C);
  gemm_bf16<0, true, false><<<dim3(C / 128, M / 128), 256, 0, stream>>>(
      hb, w2T, b2, x1, nullptr, out, M, C, Dff);
}
